// myBadTransfomerBlock_83605833384486
// MI455X (gfx1250) — matrix operand order repaired and checked by reading, not compiled, not run on hardware
//
#include <hip/hip_runtime.h>
#include <math.h>

// ---------------- problem dims (fixed by reference) ----------------
#define D     1024
#define SEQ   2048
#define BATCH 4
#define MTOT  (BATCH*SEQ)   // 8192 rows
#define FF    (4*D)         // 4096

// ---------------- WMMA types ----------------
typedef __attribute__((ext_vector_type(16))) __bf16         v16bf;
typedef __attribute__((ext_vector_type(8)))  float          v8f;
typedef __attribute__((ext_vector_type(8)))  unsigned short u16x8;

union FragBF { v16bf v; u16x8 h[2]; };

__device__ __forceinline__ unsigned short f2bf(float f) {
  unsigned int u = __float_as_uint(f);
  u += 0x7fffu + ((u >> 16) & 1u);        // round-to-nearest-even
  return (unsigned short)(u >> 16);
}
__device__ __forceinline__ unsigned int f2bf2(float lo, float hi) {
  return (unsigned)f2bf(lo) | ((unsigned)f2bf(hi) << 16);
}

// ---------------- CDNA5 async global->LDS support (probe-gated) ----------------
#if defined(__has_builtin)
#if __has_builtin(__builtin_amdgcn_global_load_async_to_lds_b128) && \
    __has_builtin(__builtin_amdgcn_s_wait_asynccnt)
#define HAVE_ASYNC_LDS 1
#endif
#endif

#ifdef HAVE_ASYNC_LDS
typedef int v4i_t __attribute__((vector_size(16)));
typedef __attribute__((address_space(1))) v4i_t* gv4i_p;
typedef __attribute__((address_space(3))) v4i_t* lv4i_p;
__device__ __forceinline__ void async_cp_b128(const void* g, void* l) {
  // flat LDS pointer = aperture | offset: low 32 bits are the LDS offset.
  gv4i_p gp = (gv4i_p)(unsigned long long)g;
  lv4i_p lp = (lv4i_p)(unsigned int)(unsigned long long)l;
  __builtin_amdgcn_global_load_async_to_lds_b128(gp, lp, 0, 0);
}
#endif

// ------- fp32 [R][C] -> bf16 [C][R] convert+transpose (64x64 LDS tiles) -------
__global__ __launch_bounds__(256)
void cvt_t_kernel(const float* __restrict__ in, unsigned short* __restrict__ out,
                  int R, int C) {
  __shared__ unsigned short tile[64 * 80];   // pad: 160B rows, 16B aligned
  const int r0 = blockIdx.y * 64, c0 = blockIdx.x * 64;
  const int t  = threadIdx.x;
  const int rr = t >> 2;            // 0..63
  const int cc = (t & 3) * 16;      // 0,16,32,48
  const float4* g = (const float4*)(in + (size_t)(r0 + rr) * C + c0 + cc);
  const float4 f0 = g[0], f1 = g[1], f2 = g[2], f3 = g[3];
  uint4 p0, p1;
  p0.x = f2bf2(f0.x, f0.y); p0.y = f2bf2(f0.z, f0.w);
  p0.z = f2bf2(f1.x, f1.y); p0.w = f2bf2(f1.z, f1.w);
  p1.x = f2bf2(f2.x, f2.y); p1.y = f2bf2(f2.z, f2.w);
  p1.z = f2bf2(f3.x, f3.y); p1.w = f2bf2(f3.z, f3.w);
  *(uint4*)(&tile[rr * 80 + cc])     = p0;
  *(uint4*)(&tile[rr * 80 + cc + 8]) = p1;
  __syncthreads();
  // out[c0+rr][r0+cc+i] = in[r0+cc+i][c0+rr] = tile[cc+i][rr]
  unsigned short v[16];
#pragma unroll
  for (int i = 0; i < 16; ++i) v[i] = tile[(cc + i) * 80 + rr];
  uint4 q0, q1;
  q0.x = v[0] | (v[1] << 16);   q0.y = v[2]  | (v[3]  << 16);
  q0.z = v[4] | (v[5] << 16);   q0.w = v[6]  | (v[7]  << 16);
  q1.x = v[8] | (v[9] << 16);   q1.y = v[10] | (v[11] << 16);
  q1.z = v[12]| (v[13] << 16);  q1.w = v[14] | (v[15] << 16);
  uint4* o = (uint4*)(out + (size_t)(c0 + rr) * R + r0 + cc);
  o[0] = q0; o[1] = q1;
}

// ------- bf16 [R][C] -> bf16 [C][R] transpose (batched, 64x64 LDS tiles) -------
__global__ __launch_bounds__(256)
void tr_bf16_kernel(const unsigned short* __restrict__ in,
                    unsigned short* __restrict__ out,
                    int R, int C, long inBatch, long outBatch) {
  __shared__ unsigned short tile[64 * 80];
  const unsigned short* ib = in  + (size_t)blockIdx.z * inBatch;
  unsigned short*       ob = out + (size_t)blockIdx.z * outBatch;
  const int r0 = blockIdx.y * 64, c0 = blockIdx.x * 64;
  const int t  = threadIdx.x;
  const int rr = t >> 2;
  const int cc = (t & 3) * 16;
  const uint4* g = (const uint4*)(ib + (size_t)(r0 + rr) * C + c0 + cc);
  const uint4 p0 = g[0], p1 = g[1];
  *(uint4*)(&tile[rr * 80 + cc])     = p0;
  *(uint4*)(&tile[rr * 80 + cc + 8]) = p1;
  __syncthreads();
  unsigned short v[16];
#pragma unroll
  for (int i = 0; i < 16; ++i) v[i] = tile[(cc + i) * 80 + rr];
  uint4 q0, q1;
  q0.x = v[0] | (v[1] << 16);   q0.y = v[2]  | (v[3]  << 16);
  q0.z = v[4] | (v[5] << 16);   q0.w = v[6]  | (v[7]  << 16);
  q1.x = v[8] | (v[9] << 16);   q1.y = v[10] | (v[11] << 16);
  q1.z = v[12]| (v[13] << 16);  q1.w = v[14] | (v[15] << 16);
  uint4* o = (uint4*)(ob + (size_t)(c0 + rr) * R + r0 + cc);
  o[0] = q0; o[1] = q1;
}

// ---------------- LayerNorm (d=1024), 256 thr/row, bf16 out ----------------
__global__ __launch_bounds__(256)
void ln_bf16_kernel(const float* __restrict__ x, const float* __restrict__ g,
                    const float* __restrict__ b, unsigned short* __restrict__ out) {
  const int row = blockIdx.x;
  const int t   = threadIdx.x;
  const float4 v = ((const float4*)(x + (size_t)row * D))[t];
  float s  = v.x + v.y + v.z + v.w;
  float sq = v.x*v.x + v.y*v.y + v.z*v.z + v.w*v.w;
  __shared__ float rs[256], rq[256];
  rs[t] = s; rq[t] = sq;
  __syncthreads();
  for (int off = 128; off > 0; off >>= 1) {
    if (t < off) { rs[t] += rs[t+off]; rq[t] += rq[t+off]; }
    __syncthreads();
  }
  const float mu   = rs[0] * (1.0f / D);
  const float var  = rq[0] * (1.0f / D) - mu * mu;
  const float rstd = rsqrtf(var + 1e-5f);
  const float4 gv = ((const float4*)g)[t];
  const float4 bv = ((const float4*)b)[t];
  uint2 pk;
  pk.x = f2bf2((v.x - mu) * rstd * gv.x + bv.x, (v.y - mu) * rstd * gv.y + bv.y);
  pk.y = f2bf2((v.z - mu) * rstd * gv.z + bv.z, (v.w - mu) * rstd * gv.w + bv.w);
  ((uint2*)(out + (size_t)row * D))[t] = pk;
}

// ---------------- softmax over SEQ=2048 cols, bf16 out ----------------
__global__ __launch_bounds__(256)
void softmax_bf16_kernel(const float* __restrict__ S, unsigned short* __restrict__ P) {
  const int row = blockIdx.x;             // 0..BATCH*SEQ-1
  const int t   = threadIdx.x;
  const float4* sr = (const float4*)(S + (size_t)row * SEQ);
  const float4 a = sr[t*2], c = sr[t*2 + 1];
  float m = fmaxf(fmaxf(fmaxf(a.x,a.y), fmaxf(a.z,a.w)),
                  fmaxf(fmaxf(c.x,c.y), fmaxf(c.z,c.w)));
  __shared__ float red[256];
  red[t] = m; __syncthreads();
  for (int off = 128; off > 0; off >>= 1) {
    if (t < off) red[t] = fmaxf(red[t], red[t+off]);
    __syncthreads();
  }
  m = red[0];
  __syncthreads();
  const float e0 = expf(a.x-m), e1 = expf(a.y-m), e2 = expf(a.z-m), e3 = expf(a.w-m);
  const float e4 = expf(c.x-m), e5 = expf(c.y-m), e6 = expf(c.z-m), e7 = expf(c.w-m);
  red[t] = e0+e1+e2+e3+e4+e5+e6+e7;
  __syncthreads();
  for (int off = 128; off > 0; off >>= 1) {
    if (t < off) red[t] += red[t+off];
    __syncthreads();
  }
  const float inv = 1.0f / red[0];
  uint4 pk;
  pk.x = f2bf2(e0*inv, e1*inv);
  pk.y = f2bf2(e2*inv, e3*inv);
  pk.z = f2bf2(e4*inv, e5*inv);
  pk.w = f2bf2(e6*inv, e7*inv);
  ((uint4*)(P + (size_t)row * SEQ))[t] = pk;
}

// ------- bf16 WMMA GEMM: out = epi(act[M,K] @ Wt^T + bias), Wt is [N][K] -------
#define BM 128
#define BN 128

#ifdef HAVE_ASYNC_LDS
// =========== async pipeline: double-buffered LDS, both operands DMA'd ===========
#define BK  32
#define LDT 48   // halfs/row: 96B rows -> 16B aligned, bank-staggered

__global__ __launch_bounds__(256)
void gemm_bf16_kernel(const unsigned short* __restrict__ Aact,
                      const unsigned short* __restrict__ Wt,
                      const float* __restrict__ bias,
                      const float* __restrict__ resid,
                      float* __restrict__ outF,
                      unsigned short* __restrict__ outB,
                      int M, int N, int K, float scale, int gelu,
                      long aBatch, long bBatch, long oBatch) {
  __shared__ unsigned short As[2][BM * LDT];
  __shared__ unsigned short Bs[2][BN * LDT];

  const int tid = threadIdx.x;
  const int z   = blockIdx.z;
  const unsigned short* Ab = Aact + (size_t)z * aBatch;
  const unsigned short* Bb = Wt   + (size_t)z * bBatch;

  const int m0 = blockIdx.y * BM;
  const int n0 = blockIdx.x * BN;

  const int wid   = tid >> 5;
  const int lane  = tid & 31;
  const int wr    = wid >> 2;     // 0..1 -> 64-row slab
  const int wc    = wid & 3;      // 0..3 -> 32-col slab
  const int mlane = lane & 15;
  const int khalf = lane >> 4;

  const v8f vzero = {0.f,0.f,0.f,0.f,0.f,0.f,0.f,0.f};
  v8f acc[4][2];
#pragma unroll
  for (int i = 0; i < 4; ++i)
#pragma unroll
    for (int j = 0; j < 2; ++j) acc[i][j] = vzero;

  // 16B-chunk staging: tile = 128 rows x 32 halfs = 512 chunks, 2 per thread
  const int crow = tid >> 2;           // 0..63 (+64 second chunk)
  const int ccol = (tid & 3) * 8;      // half offset within row

  auto issue = [&](int buf, int k0) {
#pragma unroll
    for (int i = 0; i < 2; ++i) {
      const int r = crow + i * 64;
      async_cp_b128(Ab + (size_t)(m0 + r) * K + k0 + ccol,
                    &As[buf][r * LDT + ccol]);
      async_cp_b128(Bb + (size_t)(n0 + r) * K + k0 + ccol,
                    &Bs[buf][r * LDT + ccol]);
    }
  };

  const int nk = K / BK;
  issue(0, 0);                               // prologue: fill buffer 0
  for (int kt = 0; kt < nk; ++kt) {
    const int buf = kt & 1;
    __builtin_amdgcn_s_wait_asynccnt(0);     // DMA for this buffer done
    __syncthreads();                         // buffer ready / other buffer free
    if (kt + 1 < nk) issue(buf ^ 1, (kt + 1) * BK);  // DMA next during WMMAs
    // ---- 8 WMMAs from LDS buffer `buf` ----
    FragBF a[4], b[2];
#pragma unroll
    for (int i = 0; i < 4; ++i) {
      const int r = wr * 64 + i * 16 + mlane;
      a[i].h[0] = *(const u16x8*)(&As[buf][r * LDT + khalf * 8]);
      a[i].h[1] = *(const u16x8*)(&As[buf][r * LDT + khalf * 8 + 16]);
    }
#pragma unroll
    for (int j = 0; j < 2; ++j) {
      const int c = wc * 32 + j * 16 + mlane;
      const u16x8* p = (const u16x8*)(&Bs[buf][c * LDT + khalf * 8]);
      b[j].h[0] = p[0];
      b[j].h[1] = p[2];
    }
#pragma unroll
    for (int i = 0; i < 4; ++i)
#pragma unroll
      for (int j = 0; j < 2; ++j)
        acc[i][j] = __builtin_amdgcn_wmma_f32_16x16x32_bf16(
            false, a[i].v, false, b[j].v, (short)0, acc[i][j], false, false);
  }

  // ---------------- fused epilogue ----------------
  float*          oF = outF ? outF + (size_t)z * oBatch : nullptr;
  unsigned short* oB = outB ? outB + (size_t)z * oBatch : nullptr;
#pragma unroll
  for (int i = 0; i < 4; ++i) {
    const int gmBase = m0 + wr * 64 + i * 16 + khalf * 8;
#pragma unroll
    for (int j = 0; j < 2; ++j) {
      const int gn = n0 + wc * 32 + j * 16 + mlane;
      const float bv = bias ? bias[gn] : 0.0f;
#pragma unroll
      for (int r = 0; r < 8; ++r) {
        const int gm = gmBase + r;
        float v = (acc[i][j][r] + bv) * scale;
        if (gelu) v = 0.5f * v * (1.0f + erff(v * 0.70710678118654752f));
        if (resid) v += resid[(size_t)gm * N + gn];
        const size_t idx = (size_t)gm * N + gn;
        if (oF) oF[idx] = v;
        else    oB[idx] = f2bf(v);
      }
    }
  }
}

#else  // ------------- fallback: single-buffer, batched reg staging -------------
#define BK  64
#define LDT 80

__global__ __launch_bounds__(256)
void gemm_bf16_kernel(const unsigned short* __restrict__ Aact,
                      const unsigned short* __restrict__ Wt,
                      const float* __restrict__ bias,
                      const float* __restrict__ resid,
                      float* __restrict__ outF,
                      unsigned short* __restrict__ outB,
                      int M, int N, int K, float scale, int gelu,
                      long aBatch, long bBatch, long oBatch) {
  __shared__ unsigned short As[BM * LDT];
  __shared__ unsigned short Bs[BN * LDT];

  const int tid = threadIdx.x;
  const int z   = blockIdx.z;
  const unsigned short* Ab = Aact + (size_t)z * aBatch;
  const unsigned short* Bb = Wt   + (size_t)z * bBatch;

  const int m0 = blockIdx.y * BM;
  const int n0 = blockIdx.x * BN;

  const int wid   = tid >> 5;
  const int lane  = tid & 31;
  const int wr    = wid >> 2;
  const int wc    = wid & 3;
  const int mlane = lane & 15;
  const int khalf = lane >> 4;

  const v8f vzero = {0.f,0.f,0.f,0.f,0.f,0.f,0.f,0.f};
  v8f acc[4][2];
#pragma unroll
  for (int i = 0; i < 4; ++i)
#pragma unroll
    for (int j = 0; j < 2; ++j) acc[i][j] = vzero;

  const int arow = tid >> 1;
  const int acol = (tid & 1) * 32;

  const int nk = K / BK;
  for (int kt = 0; kt < nk; ++kt) {
    const int k0 = kt * BK;
    uint4 qa[4], qb[4];
    {
      const uint4* g = (const uint4*)(Ab + (size_t)(m0 + arow) * K + k0 + acol);
      qa[0] = g[0]; qa[1] = g[1]; qa[2] = g[2]; qa[3] = g[3];
    }
    {
      const uint4* g = (const uint4*)(Bb + (size_t)(n0 + arow) * K + k0 + acol);
      qb[0] = g[0]; qb[1] = g[1]; qb[2] = g[2]; qb[3] = g[3];
    }
    __syncthreads();
    {
      uint4* s = (uint4*)(&As[arow * LDT + acol]);
      s[0] = qa[0]; s[1] = qa[1]; s[2] = qa[2]; s[3] = qa[3];
      uint4* sb = (uint4*)(&Bs[arow * LDT + acol]);
      sb[0] = qb[0]; sb[1] = qb[1]; sb[2] = qb[2]; sb[3] = qb[3];
    }
    __syncthreads();

    if (kt + 1 < nk) {
      __builtin_prefetch(Ab + (size_t)(m0 + arow) * K + k0 + BK + acol, 0, 1);
      __builtin_prefetch(Bb + (size_t)(n0 + arow) * K + k0 + BK + acol, 0, 1);
    }

#pragma unroll
    for (int kk = 0; kk < BK; kk += 32) {
      FragBF a[4], b[2];
#pragma unroll
      for (int i = 0; i < 4; ++i) {
        const int r = wr * 64 + i * 16 + mlane;
        a[i].h[0] = *(const u16x8*)(&As[r * LDT + kk + khalf * 8]);
        a[i].h[1] = *(const u16x8*)(&As[r * LDT + kk + khalf * 8 + 16]);
      }
#pragma unroll
      for (int j = 0; j < 2; ++j) {
        const int c = wc * 32 + j * 16 + mlane;
        const u16x8* p = (const u16x8*)(&Bs[c * LDT + kk + khalf * 8]);
        b[j].h[0] = p[0];
        b[j].h[1] = p[2];
      }
#pragma unroll
      for (int i = 0; i < 4; ++i)
#pragma unroll
        for (int j = 0; j < 2; ++j)
          acc[i][j] = __builtin_amdgcn_wmma_f32_16x16x32_bf16(
              false, a[i].v, false, b[j].v, (short)0, acc[i][j], false, false);
    }
  }

  float*          oF = outF ? outF + (size_t)z * oBatch : nullptr;
  unsigned short* oB = outB ? outB + (size_t)z * oBatch : nullptr;
#pragma unroll
  for (int i = 0; i < 4; ++i) {
    const int gmBase = m0 + wr * 64 + i * 16 + khalf * 8;
#pragma unroll
    for (int j = 0; j < 2; ++j) {
      const int gn = n0 + wc * 32 + j * 16 + mlane;
      const float bv = bias ? bias[gn] : 0.0f;
#pragma unroll
      for (int r = 0; r < 8; ++r) {
        const int gm = gmBase + r;
        float v = (acc[i][j][r] + bv) * scale;
        if (gelu) v = 0.5f * v * (1.0f + erff(v * 0.70710678118654752f));
        if (resid) v += resid[(size_t)gm * N + gn];
        const size_t idx = (size_t)gm * N + gn;
        if (oF) oF[idx] = v;
        else    oB[idx] = f2bf(v);
      }
    }
  }
}
#endif // HAVE_ASYNC_LDS

// ---------------- host-side orchestration ----------------
extern "C" void kernel_launch(void* const* d_in, const int* in_sizes, int n_in,
                              void* d_out, int out_size, void* d_ws, size_t ws_size,
                              hipStream_t stream) {
  (void)in_sizes; (void)n_in; (void)out_size; (void)ws_size;
  const float* x   = (const float*)d_in[0];
  const float* WA  = (const float*)d_in[1];
  const float* bA  = (const float*)d_in[2];
  const float* WB  = (const float*)d_in[3];
  const float* bB  = (const float*)d_in[4];
  const float* WC  = (const float*)d_in[5];
  const float* bC  = (const float*)d_in[6];
  const float* Wo  = (const float*)d_in[7];
  const float* bo  = (const float*)d_in[8];
  const float* g1  = (const float*)d_in[9];
  const float* be1 = (const float*)d_in[10];
  const float* g2  = (const float*)d_in[11];
  const float* be2 = (const float*)d_in[12];
  const float* W1  = (const float*)d_in[13];
  const float* b1  = (const float*)d_in[14];
  const float* W2  = (const float*)d_in[15];
  const float* b2  = (const float*)d_in[16];
  float* out = (float*)d_out;

  char* ws = (char*)d_ws;
  size_t off = 0;
  auto alloc = [&](size_t bytes) -> void* {
    off = (off + 255) & ~(size_t)255;
    void* p = ws + off;
    off += bytes;
    return p;
  };

  // transposed bf16 weights: Wt[N][K]
  unsigned short* WAt = (unsigned short*)alloc((size_t)D * D * 2);
  unsigned short* WBt = (unsigned short*)alloc((size_t)D * D * 2);
  unsigned short* WCt = (unsigned short*)alloc((size_t)D * D * 2);
  unsigned short* Wot = (unsigned short*)alloc((size_t)D * D * 2);
  unsigned short* W1t = (unsigned short*)alloc((size_t)D * FF * 2);   // [FF][D]
  unsigned short* W2t = (unsigned short*)alloc((size_t)FF * D * 2);   // [D][FF]
  unsigned short* y1bf = (unsigned short*)alloc((size_t)MTOT * D * 2);
  unsigned short* Abf  = (unsigned short*)alloc((size_t)MTOT * D * 2);
  unsigned short* Bbf  = (unsigned short*)alloc((size_t)MTOT * D * 2);
  unsigned short* Cbf  = (unsigned short*)alloc((size_t)MTOT * D * 2);
  unsigned short* Ct   = (unsigned short*)alloc((size_t)MTOT * D * 2); // [b][D][S]
  // scores f32 (64MB) is dead after softmax; reuse the region for h bf16 (64MB)
  float*          scores = (float*)alloc((size_t)BATCH * SEQ * SEQ * 4);
  unsigned short* hbf    = (unsigned short*)scores;
  unsigned short* P      = (unsigned short*)alloc((size_t)BATCH * SEQ * SEQ * 2);
  unsigned short* yattn  = (unsigned short*)alloc((size_t)MTOT * D * 2);
  float*          yres   = (float*)alloc((size_t)MTOT * D * 4);
  unsigned short* y2bf   = (unsigned short*)alloc((size_t)MTOT * D * 2);

  // 1) weights -> bf16, transposed to [N][K]
  dim3 gWsq(D / 64, D / 64, 1);
  cvt_t_kernel<<<gWsq, 256, 0, stream>>>(WA, WAt, D, D);
  cvt_t_kernel<<<gWsq, 256, 0, stream>>>(WB, WBt, D, D);
  cvt_t_kernel<<<gWsq, 256, 0, stream>>>(WC, WCt, D, D);
  cvt_t_kernel<<<gWsq, 256, 0, stream>>>(Wo, Wot, D, D);
  cvt_t_kernel<<<dim3(FF / 64, D / 64, 1), 256, 0, stream>>>(W1, W1t, D, FF);
  cvt_t_kernel<<<dim3(D / 64, FF / 64, 1), 256, 0, stream>>>(W2, W2t, FF, D);

  // 2) LN1
  ln_bf16_kernel<<<MTOT, 256, 0, stream>>>(x, g1, be1, y1bf);

  // 3) Q/K/V projections (1/sqrt(d)=1/32 folded into A)
  dim3 gProj(D / BN, MTOT / BM, 1);
  gemm_bf16_kernel<<<gProj, 256, 0, stream>>>(y1bf, WAt, bA, nullptr,
      nullptr, Abf, MTOT, D, D, 0.03125f, 0, 0, 0, 0);
  gemm_bf16_kernel<<<gProj, 256, 0, stream>>>(y1bf, WBt, bB, nullptr,
      nullptr, Bbf, MTOT, D, D, 1.0f, 0, 0, 0, 0);
  gemm_bf16_kernel<<<gProj, 256, 0, stream>>>(y1bf, WCt, bC, nullptr,
      nullptr, Cbf, MTOT, D, D, 1.0f, 0, 0, 0, 0);

  // 3b) C -> C^T per batch ([S][D] -> [D][S]) so P@C uses the async path too
  tr_bf16_kernel<<<dim3(D / 64, SEQ / 64, BATCH), 256, 0, stream>>>(
      Cbf, Ct, SEQ, D, (long)SEQ * D, (long)SEQ * D);

  // 4) scores = A @ B^T  (operand already [N][K])
  dim3 gScore(SEQ / BN, SEQ / BM, BATCH);
  gemm_bf16_kernel<<<gScore, 256, 0, stream>>>(Abf, Bbf, nullptr, nullptr,
      scores, nullptr, SEQ, SEQ, D, 1.0f, 0,
      (long)SEQ * D, (long)SEQ * D, (long)SEQ * SEQ);

  // 5) softmax -> P (bf16)
  softmax_bf16_kernel<<<MTOT, 256, 0, stream>>>(scores, P);

  // 6) yattn = P @ C   (Wt = C^T [D][S])
  dim3 gAV(D / BN, SEQ / BM, BATCH);
  gemm_bf16_kernel<<<gAV, 256, 0, stream>>>(P, Ct, nullptr, nullptr,
      nullptr, yattn, SEQ, D, SEQ, 1.0f, 0,
      (long)SEQ * SEQ, (long)D * SEQ, (long)SEQ * D);

  // 7) O-projection + residual(x) -> yres (f32)
  gemm_bf16_kernel<<<gProj, 256, 0, stream>>>(yattn, Wot, bo, x,
      yres, nullptr, MTOT, D, D, 1.0f, 0, 0, 0, 0);

  // 8) LN2
  ln_bf16_kernel<<<MTOT, 256, 0, stream>>>(yres, g2, be2, y2bf);

  // 9) MLP up + exact GELU -> h (bf16, reuses scores region)
  dim3 gFF1(FF / BN, MTOT / BM, 1);
  gemm_bf16_kernel<<<gFF1, 256, 0, stream>>>(y2bf, W1t, b1, nullptr,
      nullptr, hbf, MTOT, FF, D, 1.0f, 1, 0, 0, 0);

  // 10) MLP down + residual(x, per reference's literal `return y + x`) -> out
  dim3 gFF2(D / BN, MTOT / BM, 1);
  gemm_bf16_kernel<<<gFF2, 256, 0, stream>>>(hbf, W2t, b2, x,
      out, nullptr, MTOT, D, FF, 1.0f, 0, 0, 0, 0);
}
